// AnatomyNet_1537598292394
// MI455X (gfx1250) — matrix operand order repaired and checked by reading, not compiled, not run on hardware
//
#include <hip/hip_runtime.h>
#include <hip/hip_bf16.h>

// ---------------- problem constants ----------------
#define Bsz   2
#define Cch   32
#define DMv   64
#define S3    262144          // 64^3
#define NROI  90
#define HIDs  64
#define REMBs 32
#define EPSv  1e-5f

typedef __bf16 bf16_t;
typedef __bf16        v16bf __attribute__((ext_vector_type(16)));
typedef float         v8f   __attribute__((ext_vector_type(8)));
typedef unsigned int  v8u   __attribute__((ext_vector_type(8)));
typedef unsigned int  v4u   __attribute__((ext_vector_type(4)));
typedef int           v4i   __attribute__((ext_vector_type(4)));

#if defined(__has_builtin)
#if __has_builtin(__builtin_amdgcn_global_load_async_to_lds_b128)
#define HAVE_ASYNC_LDS 1
#endif
#endif

// f32 -> bf16 round-to-nearest-even (as raw u16)
__device__ __forceinline__ unsigned short f2bfu(float f) {
    union { float f; unsigned u; } x; x.f = f;
    unsigned r = x.u + 0x7FFFu + ((x.u >> 16) & 1u);
    return (unsigned short)(r >> 16);
}
__device__ __forceinline__ unsigned pack2(float lo, float hi) {
    return (unsigned)f2bfu(lo) | ((unsigned)f2bfu(hi) << 16);
}

// ---------------------------------------------------------------------------
// Weight prep: convs_w f32 (3,Co,Ci,3,3,3) -> bf16 (3,tap=27,Co,Ci), Ci contiguous
// so B-fragments are contiguous 32B per lane.
// ---------------------------------------------------------------------------
__global__ __launch_bounds__(256) void prep_wt_kernel(
    const float* __restrict__ w, unsigned short* __restrict__ wt) {
    int i = blockIdx.x * 256 + threadIdx.x;            // dest index
    const int total = 3 * 27 * 32 * 32;
    if (i >= total) return;
    int ci  = i & 31;
    int co  = (i >> 5) & 31;
    int tap = (i >> 10) % 27;
    int l   = (i >> 10) / 27;
    float f = w[(((l * 32 + co) * 32 + ci) * 27) + tap];
    wt[i] = f2bfu(f);
}

// ---------------------------------------------------------------------------
// Layer 0: 1 -> 32 channels, K=27 — VALU. Writes y channel-last (b,s,c) f32
// and accumulates per-(b,c) sum / sumsq for InstanceNorm.
// ---------------------------------------------------------------------------
__global__ __launch_bounds__(256) void conv0_kernel(
    const float* __restrict__ data, const float* __restrict__ w0,
    const float* __restrict__ b0, float* __restrict__ y,
    float* __restrict__ stats) {
    __shared__ float wl[32 * 27];
    __shared__ float bl[32];
    __shared__ float statS[32], statQ[32];
    const int tid = threadIdx.x;
    for (int i = tid; i < 32 * 27; i += 256) wl[i] = w0[i];
    if (tid < 32) { bl[tid] = b0[tid]; statS[tid] = 0.f; statQ[tid] = 0.f; }
    __syncthreads();

    long gid = (long)blockIdx.x * 256 + tid;           // b*S + s
    int b = (int)(gid >> 18);
    int s = (int)(gid & (S3 - 1));
    int z = s >> 12, yy = (s >> 6) & 63, x = s & 63;

    float v[27];
    int t = 0;
    #pragma unroll
    for (int dz = -1; dz <= 1; ++dz)
        #pragma unroll
        for (int dy = -1; dy <= 1; ++dy)
            #pragma unroll
            for (int dx = -1; dx <= 1; ++dx) {
                int zz = z + dz, yv = yy + dy, xx = x + dx;
                bool inb = (unsigned)zz < 64u && (unsigned)yv < 64u && (unsigned)xx < 64u;
                long off = (long)b * S3 + (long)zz * 4096 + yv * 64 + xx;
                v[t++] = inb ? data[off] : 0.f;
            }
    for (int c = 0; c < 32; ++c) {
        float acc = bl[c];
        #pragma unroll
        for (int k = 0; k < 27; ++k) acc = fmaf(wl[c * 27 + k], v[k], acc);
        y[gid * 32 + c] = acc;
        atomicAdd(&statS[c], acc);
        atomicAdd(&statQ[c], acc * acc);
    }
    __syncthreads();
    int bb = (blockIdx.x * 256) >> 18;
    if (tid < 32) {
        atomicAdd(&stats[(bb * 2) * 32 + tid],     statS[tid]);
        atomicAdd(&stats[(bb * 2 + 1) * 32 + tid], statQ[tid]);
    }
}

// ---------------------------------------------------------------------------
// InstanceNorm + ReLU + bf16 re-quant, channel-last.
// ---------------------------------------------------------------------------
__global__ __launch_bounds__(256) void norm_relu_bf16_kernel(
    const float* __restrict__ y, const float* __restrict__ stats,
    unsigned short* __restrict__ xb) {
    __shared__ float meanL[32], rsqL[32];
    const int tid = threadIdx.x;
    long gid = (long)blockIdx.x * 256 + tid;
    int b = (int)(gid >> 18);
    if (tid < 32) {
        float m = stats[(b * 2) * 32 + tid] * (1.0f / S3);
        float q = stats[(b * 2 + 1) * 32 + tid] * (1.0f / S3);
        meanL[tid] = m;
        rsqL[tid]  = rsqrtf(q - m * m + EPSv);
    }
    __syncthreads();
    const float4* yp = (const float4*)(y + gid * 32);
    unsigned ow[16];
    #pragma unroll
    for (int i = 0; i < 8; ++i) {
        float4 f = yp[i];
        int c = i * 4;
        float a0 = fmaxf((f.x - meanL[c + 0]) * rsqL[c + 0], 0.f);
        float a1 = fmaxf((f.y - meanL[c + 1]) * rsqL[c + 1], 0.f);
        float a2 = fmaxf((f.z - meanL[c + 2]) * rsqL[c + 2], 0.f);
        float a3 = fmaxf((f.w - meanL[c + 3]) * rsqL[c + 3], 0.f);
        ow[i * 2]     = pack2(a0, a1);
        ow[i * 2 + 1] = pack2(a2, a3);
    }
    uint4* op = (uint4*)(xb + gid * 32);
    #pragma unroll
    for (int i = 0; i < 4; ++i)
        op[i] = make_uint4(ow[i * 4], ow[i * 4 + 1], ow[i * 4 + 2], ow[i * 4 + 3]);
}

// ---------------------------------------------------------------------------
// 32->32 3x3x3 conv as implicit GEMM with v_wmma_f32_16x16x32_bf16.
// Wave tile: 16 voxels (along x) x 32 out channels. 2 WMMAs per tap, 27 taps.
// Weights staged in LDS (54KB) via CDNA5 async global->LDS b128 when available.
// Halo handled branchlessly (clamped addr + mask) so EXEC stays all-ones
// around the WMMAs.
// ---------------------------------------------------------------------------
__global__ __launch_bounds__(256) void conv_wmma_kernel(
    const unsigned short* __restrict__ xb,  // (B,S,32) bf16
    const unsigned short* __restrict__ wt,  // (27,32,32) bf16, this layer
    const float* __restrict__ bias,         // (32)
    float* __restrict__ y,                  // (B,S,32) f32
    float* __restrict__ stats) {
    __shared__ unsigned short wlds[27 * 32 * 32];
    __shared__ float statS[32], statQ[32];
    const int tid = threadIdx.x;

    // Stage 55296B of weights: 3456 x 16-byte chunks.
#if defined(HAVE_ASYNC_LDS)
    {
        typedef __attribute__((address_space(1))) v4i* gv4p;
        typedef __attribute__((address_space(3))) v4i* lv4p;
        for (int i = tid; i < 3456; i += 256) {
            __builtin_amdgcn_global_load_async_to_lds_b128(
                (gv4p)(unsigned long long)(const void*)(wt + i * 8),
                (lv4p)(unsigned long long)(void*)(&wlds[i * 8]), 0, 0);
        }
#if __has_builtin(__builtin_amdgcn_s_wait_asynccnt)
        __builtin_amdgcn_s_wait_asynccnt(0);
#else
        __asm__ volatile("s_wait_asynccnt 0" ::: "memory");
#endif
    }
#else
    {
        const uint4* src = (const uint4*)wt;
        uint4* dst = (uint4*)wlds;
        for (int i = tid; i < 3456; i += 256) dst[i] = src[i];
    }
#endif
    if (tid < 32) { statS[tid] = 0.f; statQ[tid] = 0.f; }
    __syncthreads();

    const int wave  = tid >> 5;
    const int lane  = tid & 31;
    const int mrow  = lane & 15;    // voxel-in-tile for A rows / N column for B,C,D
    const int khalf = lane >> 4;    // lane half (K sub-range select)

    int tile = blockIdx.x * 8 + wave;       // 16-voxel tile along x
    int x0   = (tile & 3) << 4;
    int row  = tile >> 2;                   // b*4096 + z*64 + y
    int b    = row >> 12;
    int z    = (row >> 6) & 63;
    int yy0  = row & 63;
    const long base_b = (long)b * S3;

    v8f c0 = {}; v8f c1 = {};

    for (int t = 0; t < 27; ++t) {
        int dz = t / 9, dy = (t / 3) % 3, dx = t % 3;
        int zz = z + dz - 1, yv = yy0 + dy - 1;
        if ((unsigned)zz >= 64u || (unsigned)yv >= 64u) continue;  // wave-uniform

        int xx   = x0 + mrow + dx - 1;
        bool inb = (unsigned)xx < 64u;
        long vbase = (base_b + (long)zz * 4096 + (long)yv * 64 + (inb ? xx : 0)) * 32;

        // A fragment, this code's k order (not the ISA table order): lane half h holds K in
        // {16h..16h+15} (contiguous, matching B)
        v4u lo = *(const v4u*)(xb + vbase + khalf * 16);
        v4u hi = *(const v4u*)(xb + vbase + khalf * 16 + 8);
        unsigned msk = inb ? 0xFFFFFFFFu : 0u;
        v8u raw;
        #pragma unroll
        for (int i = 0; i < 4; ++i) { raw[i] = lo[i] & msk; raw[4 + i] = hi[i] & msk; }
        union { v8u u; v16bf b; } cv; cv.u = raw;
        v16bf afrag = cv.b;

        // B fragments: column N = n*16 + mrow, K range = khalf*16..+15 (contiguous)
        const unsigned short* wp = &wlds[((t * 32 + mrow) * 32) + khalf * 16];
        v16bf b0 = *(const v16bf*)wp;
        v16bf b1 = *(const v16bf*)(wp + 16 * 32);

        c0 = __builtin_amdgcn_wmma_f32_16x16x32_bf16(false, afrag, false, b0,
                                                     (short)0, c0, false, false);
        c1 = __builtin_amdgcn_wmma_f32_16x16x32_bf16(false, afrag, false, b1,
                                                     (short)0, c1, false, false);
    }

    // Bias, store channel-last, per-channel partial stats.
    float bia0 = bias[mrow], bia1 = bias[16 + mrow];
    float s0 = 0.f, q0 = 0.f, s1 = 0.f, q1 = 0.f;
    long obase = (base_b + (long)z * 4096 + (long)yy0 * 64 + x0) * 32;
    #pragma unroll
    for (int v = 0; v < 8; ++v) {
        int m = v + khalf * 8;                 // C/D layout: M = v + 8*(lane/16)
        float y0v = c0[v] + bia0;
        float y1v = c1[v] + bia1;
        y[obase + (long)m * 32 + mrow]      = y0v;
        y[obase + (long)m * 32 + 16 + mrow] = y1v;
        s0 += y0v; q0 += y0v * y0v;
        s1 += y1v; q1 += y1v * y1v;
    }
    atomicAdd(&statS[mrow], s0);      atomicAdd(&statQ[mrow], q0);
    atomicAdd(&statS[16 + mrow], s1); atomicAdd(&statQ[16 + mrow], q1);
    __syncthreads();
    int bb = (blockIdx.x * 8) >> 14;  // block-uniform batch index
    if (tid < 32) {
        atomicAdd(&stats[(bb * 2) * 32 + tid],     statS[tid]);
        atomicAdd(&stats[(bb * 2 + 1) * 32 + tid], statQ[tid]);
    }
}

// ---------------------------------------------------------------------------
// ROI pooling with the final InstanceNorm+ReLU fused into the read.
// One pass over atlas (94MB) and features (67MB); LDS accumulators per block.
// ---------------------------------------------------------------------------
__global__ __launch_bounds__(256) void pool_kernel(
    const float* __restrict__ y, const float* __restrict__ stats,
    const float* __restrict__ atlas, float* __restrict__ roi_raw,
    float* __restrict__ cnt) {
    __shared__ float acc[NROI * 32];
    __shared__ float cl[NROI];
    __shared__ float meanL[32], rsqL[32];
    const int tid = threadIdx.x;
    const int b = blockIdx.y;
    for (int i = tid; i < NROI * 32; i += 256) acc[i] = 0.f;
    if (tid < NROI) cl[tid] = 0.f;
    if (tid < 32) {
        float m = stats[(b * 2) * 32 + tid] * (1.0f / S3);
        float q = stats[(b * 2 + 1) * 32 + tid] * (1.0f / S3);
        meanL[tid] = m;
        rsqL[tid]  = rsqrtf(q - m * m + EPSv);
    }
    __syncthreads();

    for (int k = 0; k < 8; ++k) {
        int s = blockIdx.x * 2048 + k * 256 + tid;
        const float4* ep = (const float4*)(y + ((long)b * S3 + s) * 32);
        float e[32];
        #pragma unroll
        for (int i = 0; i < 8; ++i) {
            float4 f = ep[i];
            int c = i * 4;
            e[c + 0] = fmaxf((f.x - meanL[c + 0]) * rsqL[c + 0], 0.f);
            e[c + 1] = fmaxf((f.y - meanL[c + 1]) * rsqL[c + 1], 0.f);
            e[c + 2] = fmaxf((f.z - meanL[c + 2]) * rsqL[c + 2], 0.f);
            e[c + 3] = fmaxf((f.w - meanL[c + 3]) * rsqL[c + 3], 0.f);
        }
        for (int r = 0; r < NROI; ++r) {
            // hint the next atlas row (1MB stride) into cache
            if (r + 1 < NROI)
                __builtin_prefetch(&atlas[(long)(r + 1) * S3 + s], 0, 1);
            float m = atlas[(long)r * S3 + s];
            if (m > 0.5f) {
                #pragma unroll
                for (int c = 0; c < 32; ++c) atomicAdd(&acc[r * 32 + c], e[c]);
                if (b == 0) atomicAdd(&cl[r], 1.0f);
            }
        }
    }
    __syncthreads();
    for (int i = tid; i < NROI * 32; i += 256)
        atomicAdd(&roi_raw[(long)b * NROI * 32 + i], acc[i]);
    if (b == 0 && tid < NROI) atomicAdd(&cnt[tid], cl[tid]);
}

// ---------------------------------------------------------------------------
// Per-ROI gated MLP experts (tiny). One block per ROI.
// ---------------------------------------------------------------------------
__global__ __launch_bounds__(64) void mlp_kernel(
    const float* __restrict__ roi_raw, const float* __restrict__ cnt,
    const float* __restrict__ sw1, const float* __restrict__ sb1,
    const float* __restrict__ sw2, const float* __restrict__ sb2,
    const float* __restrict__ pw1, const float* __restrict__ pb1,
    const float* __restrict__ pw2, const float* __restrict__ pb2,
    float* __restrict__ out) {
    __shared__ float roiL[32], hL[64], fL[32], pL[64];
    const int t = threadIdx.x;
    const int r = blockIdx.x;
    const float inv = 1.0f / cnt[r];
    for (int b = 0; b < Bsz; ++b) {
        if (t < 32) roiL[t] = roi_raw[(b * NROI + r) * 32 + t] * inv;
        __syncthreads();
        {
            float a = sb1[r * 64 + t];
            for (int c = 0; c < 32; ++c) a = fmaf(roiL[c], sw1[(r * 32 + c) * 64 + t], a);
            hL[t] = fmaxf(a, 0.f);
        }
        __syncthreads();
        if (t < 32) {
            float a = sb2[r * 32 + t];
            for (int h = 0; h < 64; ++h) a = fmaf(hL[h], sw2[(r * 64 + h) * 32 + t], a);
            float g = 1.0f / (1.0f + __expf(-a));
            fL[t] = g * roiL[t];
        }
        __syncthreads();
        {
            float a = pb1[r * 64 + t];
            for (int c = 0; c < 32; ++c) a = fmaf(fL[c], pw1[(r * 32 + c) * 64 + t], a);
            pL[t] = fmaxf(a, 0.f);
        }
        __syncthreads();
        if (t < 32) {
            float a = pb2[r * 32 + t];
            for (int h = 0; h < 64; ++h) a = fmaf(pL[h], pw2[(r * 64 + h) * 32 + t], a);
            out[(b * NROI + r) * 32 + t] = a;
        }
        __syncthreads();
    }
}

// ---------------------------------------------------------------------------
// Host-side orchestration (all on `stream`; only graph-capturable calls).
// ---------------------------------------------------------------------------
extern "C" void kernel_launch(void* const* d_in, const int* in_sizes, int n_in,
                              void* d_out, int out_size, void* d_ws, size_t ws_size,
                              hipStream_t stream) {
    const float* data    = (const float*)d_in[0];
    const float* atlas   = (const float*)d_in[1];
    const float* conv0_w = (const float*)d_in[2];
    const float* conv0_b = (const float*)d_in[3];
    const float* convs_w = (const float*)d_in[4];
    const float* convs_b = (const float*)d_in[5];
    const float* sw1 = (const float*)d_in[6];
    const float* sb1 = (const float*)d_in[7];
    const float* sw2 = (const float*)d_in[8];
    const float* sb2 = (const float*)d_in[9];
    const float* pw1 = (const float*)d_in[10];
    const float* pb1 = (const float*)d_in[11];
    const float* pw2 = (const float*)d_in[12];
    const float* pb2 = (const float*)d_in[13];

    char* ws = (char*)d_ws;
    const size_t offY   = 0;                                   // B*S*32 f32
    const size_t offXB  = offY  + (size_t)Bsz * S3 * 32 * 4;   // B*S*32 bf16
    const size_t offWT  = offXB + (size_t)Bsz * S3 * 32 * 2;   // 3*27*32*32 bf16
    const size_t offST  = offWT + (size_t)3 * 27 * 32 * 32 * 2;// B*2*32 f32
    const size_t offROI = offST + 1024;                        // B*NROI*32 f32
    const size_t offCNT = offROI + (size_t)Bsz * NROI * 32 * 4;// NROI f32

    float*          y        = (float*)(ws + offY);
    unsigned short* xb       = (unsigned short*)(ws + offXB);
    unsigned short* wt       = (unsigned short*)(ws + offWT);
    float*          stats    = (float*)(ws + offST);
    float*          roi_raw  = (float*)(ws + offROI);
    float*          cntb     = (float*)(ws + offCNT);
    float*          outp     = (float*)d_out;

    prep_wt_kernel<<<(3 * 27 * 1024 + 255) / 256, 256, 0, stream>>>(convs_w, wt);

    const int nvox_blocks = (Bsz * S3) / 256;                  // 2048
    (void)hipMemsetAsync(stats, 0, Bsz * 2 * 32 * sizeof(float), stream);
    conv0_kernel<<<nvox_blocks, 256, 0, stream>>>(data, conv0_w, conv0_b, y, stats);
    norm_relu_bf16_kernel<<<nvox_blocks, 256, 0, stream>>>(y, stats, xb);

    for (int l = 0; l < 3; ++l) {
        (void)hipMemsetAsync(stats, 0, Bsz * 2 * 32 * sizeof(float), stream);
        conv_wmma_kernel<<<(Bsz * 4096 * 4) / 8, 256, 0, stream>>>(
            xb, wt + (size_t)l * 27 * 1024, convs_b + l * 32, y, stats);
        if (l < 2)
            norm_relu_bf16_kernel<<<nvox_blocks, 256, 0, stream>>>(y, stats, xb);
    }

    (void)hipMemsetAsync(roi_raw, 0, Bsz * NROI * 32 * sizeof(float), stream);
    (void)hipMemsetAsync(cntb, 0, NROI * sizeof(float), stream);
    dim3 pg(S3 / 2048, Bsz);
    pool_kernel<<<pg, 256, 0, stream>>>(y, stats, atlas, roi_raw, cntb);

    mlp_kernel<<<NROI, 64, 0, stream>>>(roi_raw, cntb, sw1, sb1, sw2, sb2,
                                        pw1, pb1, pw2, pb2, outp);
}
